// PatchedMixtralSparseMoeBlock_48249662603723
// MI455X (gfx1250) — matrix operand order repaired and checked by reading, not compiled, not run on hardware
//
#include <hip/hip_runtime.h>
#include <hip/hip_bf16.h>
#include <math.h>

// ---- problem constants (from reference) ----
#define TT   4096          // tokens = B*S
#define HH   1024          // hidden
#define FF   4096          // ffn dim
#define EE   8             // experts
#define XSTRIDE 1032       // effective LDS row stride (bf16) incl. TDM padding
#define GSTRIDE 68         // LDS row stride (f32) for gate/up exchange
#define ASTRIDE 72         // LDS row stride (bf16) for post-silu A tile

typedef __attribute__((ext_vector_type(16))) __bf16 v16bf;
typedef __attribute__((ext_vector_type(8)))  __bf16 v8bf;
typedef __attribute__((ext_vector_type(8)))  float  v8f;
typedef __attribute__((ext_vector_type(4)))  unsigned int v4u;
typedef __attribute__((ext_vector_type(8)))  int v8i;
typedef __attribute__((ext_vector_type(4)))  int v4i;

#if defined(__has_builtin)
#if __has_builtin(__builtin_amdgcn_tensor_load_to_lds) && __has_builtin(__builtin_amdgcn_s_wait_tensorcnt)
#define HAVE_TDM 1
#endif
#endif

// ---------------------------------------------------------------------------
// Kernel 0: zero the atomic-accumulated output region and the expert counters
// ---------------------------------------------------------------------------
__global__ __launch_bounds__(256) void moe_zero(float* __restrict__ out,
                                                int* __restrict__ counts) {
    size_t i = (size_t)blockIdx.x * blockDim.x + threadIdx.x;   // 4096*256 = T*H/4
    ((float4*)out)[i] = make_float4(0.f, 0.f, 0.f, 0.f);
    if (blockIdx.x == 0 && threadIdx.x < EE) counts[threadIdx.x] = 0;
}

// ---------------------------------------------------------------------------
// Kernel 1: router. One wave32 per token: logits = x @ gate^T, softmax, top-2,
// append (token, weight) to expert buckets, convert x row to bf16 workspace.
// ---------------------------------------------------------------------------
__global__ __launch_bounds__(256) void moe_router(const float* __restrict__ x,
                                                  const float* __restrict__ gw,
                                                  float* __restrict__ logits,
                                                  __bf16* __restrict__ xbf,
                                                  int* __restrict__ counts,
                                                  int* __restrict__ etok,
                                                  float* __restrict__ ewt) {
    int lane  = threadIdx.x & 31;
    int token = blockIdx.x * (blockDim.x >> 5) + (threadIdx.x >> 5);
    if (token >= TT) return;

    float acc[EE];
#pragma unroll
    for (int e = 0; e < EE; ++e) acc[e] = 0.f;

    const float* xr = x + (size_t)token * HH;
    for (int h = lane; h < HH; h += 32) {
        float xv = xr[h];
        xbf[(size_t)token * HH + h] = (__bf16)xv;
#pragma unroll
        for (int e = 0; e < EE; ++e) acc[e] += xv * gw[e * HH + h];
    }
#pragma unroll
    for (int off = 16; off > 0; off >>= 1) {
#pragma unroll
        for (int e = 0; e < EE; ++e) acc[e] += __shfl_xor(acc[e], off, 32);
    }

    if (lane == 0) {
#pragma unroll
        for (int e = 0; e < EE; ++e) logits[token * EE + e] = acc[e];
        float mx = acc[0];
#pragma unroll
        for (int e = 1; e < EE; ++e) mx = fmaxf(mx, acc[e]);
        float ex[EE];
        float s = 0.f;
#pragma unroll
        for (int e = 0; e < EE; ++e) { ex[e] = __expf(acc[e] - mx); s += ex[e]; }
        int i0 = 0;
#pragma unroll
        for (int e = 1; e < EE; ++e) if (ex[e] > ex[i0]) i0 = e;
        int i1 = (i0 == 0) ? 1 : 0;
#pragma unroll
        for (int e = 0; e < EE; ++e) if (e != i0 && ex[e] > ex[i1]) i1 = e;
        float denom = ex[i0] + ex[i1];
        float w0 = ex[i0] / denom, w1 = ex[i1] / denom;

        int p0 = atomicAdd(&counts[i0], 1);
        etok[i0 * TT + p0] = token;  ewt[i0 * TT + p0] = w0;
        int p1 = atomicAdd(&counts[i1], 1);
        etok[i1 * TT + p1] = token;  ewt[i1 * TT + p1] = w1;
    }
}

// ---------------------------------------------------------------------------
// Kernel 2: fused expert FFN via WMMA bf16, X tile gathered by the TDM
// (tensor_load_to_lds, gather mode: 16 x 16-bit row indices, zero-fill OOB).
// ---------------------------------------------------------------------------
__global__ __launch_bounds__(256) void moe_ffn(const __bf16* __restrict__ xbf,
                                               const float*  __restrict__ w13,
                                               const float*  __restrict__ w2,
                                               const int*    __restrict__ counts,
                                               const int*    __restrict__ etok,
                                               const float*  __restrict__ ewt,
                                               float* __restrict__ out) {
    const int e  = blockIdx.x;
    const int mt = blockIdx.y;
    const int cnt = counts[e];
    if (mt * 16 >= cnt) return;

    __shared__ __bf16 Xs[16 * XSTRIDE];     // gathered token rows, bf16 (TDM dest)
    __shared__ float  Gs[16 * GSTRIDE];
    __shared__ float  Us[16 * GSTRIDE];
    __shared__ __bf16 As[16 * ASTRIDE];
    __shared__ int    toks[16];
    __shared__ float  wts[16];

    const int tid  = threadIdx.x;
    const int wv   = tid >> 5;
    const int lane = tid & 31;
    const int nl   = lane & 15;
    const int half = lane >> 4;

    if (tid < 16) {
        int slot = mt * 16 + tid;
        int tk   = (slot < cnt) ? etok[e * TT + slot] : -1;
        toks[tid] = tk;
        wts[tid]  = (slot < cnt) ? ewt[e * TT + slot] : 0.f;
    }

#ifdef HAVE_TDM
    if (wv == 0) {
        // Build gather-mode D# (all dwords forced uniform via readfirstlane).
        unsigned idxw[8];
#pragma unroll
        for (int p = 0; p < 8; ++p) {
            int s0 = mt * 16 + 2 * p, s1 = s0 + 1;
            unsigned lo = (s0 < cnt) ? (unsigned)etok[e * TT + s0] : 0xFFFFu; // OOB -> zero-fill
            unsigned hi = (s1 < cnt) ? (unsigned)etok[e * TT + s1] : 0xFFFFu;
            idxw[p] = (unsigned)__builtin_amdgcn_readfirstlane((int)(lo | (hi << 16)));
        }
        unsigned lds_addr = (unsigned)__builtin_amdgcn_readfirstlane(
                                (int)(unsigned)(uintptr_t)(&Xs[0]));
        unsigned long long ga = (unsigned long long)(uintptr_t)xbf;
        unsigned ga_lo = (unsigned)__builtin_amdgcn_readfirstlane((int)(unsigned)ga);
        unsigned ga_hi = (unsigned)__builtin_amdgcn_readfirstlane(
                             (int)(((unsigned)(ga >> 32)) & 0x01FFFFFFu));

        // group0: count=1 | gather_mode | lds_addr | global_addr | type=2
        v4u g0 = { 0x80000001u, lds_addr, ga_lo, ga_hi | 0x80000000u };
        // group1: data_size=2B, pad_enable, pad_interval=256dw, pad_amount=2dw,
        //         tensor_dim0=1024, tensor_dim1=4096, tile_dim0=1024,
        //         tile_dim1=16 indices, tensor_dim0_stride=1024
        v8i g1 = { (int)0x03D10000u,        // mask=0,ds=1,pad_en,int=7,amt=1
                   (int)0x04000000u,        // tensor_dim0[15:0]=1024 in [31:16]
                   (int)0x10000000u,        // tensor_dim1[15:0]=4096 in [31:16]
                   (int)0x04000000u,        // tile_dim0=1024 in [31:16]
                   (int)0x00000010u,        // tile_dim1=16
                   (int)1024,               // tensor_dim0_stride lo
                   0, 0 };
        v4i g2 = { (int)idxw[0], (int)idxw[1], (int)idxw[2], (int)idxw[3] };
        v4i g3 = { (int)idxw[4], (int)idxw[5], (int)idxw[6], (int)idxw[7] };
        v8i g23 = { (int)idxw[0], (int)idxw[1], (int)idxw[2], (int)idxw[3],
                    (int)idxw[4], (int)idxw[5], (int)idxw[6], (int)idxw[7] };
        __builtin_amdgcn_tensor_load_to_lds(g0, g1, g2, g3, g23, 0);
        __builtin_amdgcn_s_wait_tensorcnt(0);
    }
#else
    __syncthreads();
    {
        int r = tid >> 4;
        int c = (tid & 15) * 64;
        int tk = toks[r];
#pragma unroll
        for (int j = 0; j < 8; ++j) {
            int cc = c + j * 8;
            v8bf v;
            if (tk >= 0) v = *(const v8bf*)(xbf + (size_t)tk * HH + cc);
            else {
#pragma unroll
                for (int q = 0; q < 8; ++q) v[q] = (__bf16)0.f;
            }
            *(v8bf*)(&Xs[r * 1032 + cc + ((cc >= 512) ? 4 : 0)]) = v;
        }
    }
#endif
    __syncthreads();

    v8f acc2[8];
#pragma unroll
    for (int t = 0; t < 8; ++t) acc2[t] = (v8f)(0.f);

    const float* wb13 = w13 + (size_t)e * 2 * FF * HH;
    const float* wb2  = w2  + (size_t)e * HH * FF;

    for (int fc = 0; fc < FF / 64; ++fc) {
        // ---------------- GEMM1: 16 x 64 (gate or up), K over H ----------------
        const int isUp = wv >> 2;
        const int ncol = fc * 64 + (wv & 3) * 16;
        const float* wrp = wb13 + (size_t)((isUp ? FF : 0) + ncol + nl) * HH + half * 16;

        v8f acc1 = (v8f)(0.f);
        for (int kk = 0; kk < HH; kk += 32) {
            __builtin_prefetch(wrp + kk + 128, 0, 3);
            int koff = kk + ((kk >= 512) ? 4 : 0);           // TDM mid-row pad skip
            v8bf a0 = *(const v8bf*)(&Xs[nl * XSTRIDE + koff + half * 16]);
            v8bf a1 = *(const v8bf*)(&Xs[nl * XSTRIDE + koff + half * 16 + 8]);
            v16bf a;
#pragma unroll
            for (int j = 0; j < 8; ++j) { a[j] = a0[j]; a[8 + j] = a1[j]; }
            float4 b0 = *(const float4*)(wrp + kk);
            float4 b1 = *(const float4*)(wrp + kk + 4);
            float4 b2 = *(const float4*)(wrp + kk + 8);
            float4 b3 = *(const float4*)(wrp + kk + 12);
            v16bf b;
            b[0]=(__bf16)b0.x; b[1]=(__bf16)b0.y; b[2]=(__bf16)b0.z; b[3]=(__bf16)b0.w;
            b[4]=(__bf16)b1.x; b[5]=(__bf16)b1.y; b[6]=(__bf16)b1.z; b[7]=(__bf16)b1.w;
            b[8]=(__bf16)b2.x; b[9]=(__bf16)b2.y; b[10]=(__bf16)b2.z; b[11]=(__bf16)b2.w;
            b[12]=(__bf16)b3.x; b[13]=(__bf16)b3.y; b[14]=(__bf16)b3.z; b[15]=(__bf16)b3.w;

            acc1 = __builtin_amdgcn_wmma_f32_16x16x32_bf16(
                       false, a, false, b, (short)0, acc1, false, false);
        }
        {
            float* dst = isUp ? Us : Gs;
#pragma unroll
            for (int r = 0; r < 8; ++r) {
                int m = r + half * 8;
                dst[m * GSTRIDE + (wv & 3) * 16 + nl] = acc1[r];
            }
        }
        __syncthreads();

        // ---------------- silu(g) * u -> bf16 A tile (fast rcp) ----------------
        {
            int base = tid * 4;
#pragma unroll
            for (int q = 0; q < 4; ++q) {
                int idx = base + q;
                int m = idx >> 6, c = idx & 63;
                float g = Gs[m * GSTRIDE + c];
                float u = Us[m * GSTRIDE + c];
                float sg = __builtin_amdgcn_rcpf(1.f + __expf(-g));
                As[m * ASTRIDE + c] = (__bf16)(g * sg * u);
            }
        }
        __syncthreads();

        // ---------------- GEMM2: out(16x1024) += A(16x64) @ W2chunk^T ----------------
#pragma unroll
        for (int t = 0; t < 8; ++t) {
            int n0 = (wv * 8 + t) * 16;
            const float* w2p = wb2 + (size_t)(n0 + nl) * FF + fc * 64 + half * 16;
#pragma unroll
            for (int k2 = 0; k2 < 2; ++k2) {
                v8bf a0 = *(const v8bf*)(&As[nl * ASTRIDE + k2 * 32 + half * 16]);
                v8bf a1 = *(const v8bf*)(&As[nl * ASTRIDE + k2 * 32 + half * 16 + 8]);
                v16bf a;
#pragma unroll
                for (int j = 0; j < 8; ++j) { a[j] = a0[j]; a[8 + j] = a1[j]; }
                float4 b0 = *(const float4*)(w2p + k2 * 32);
                float4 b1 = *(const float4*)(w2p + k2 * 32 + 4);
                float4 b2 = *(const float4*)(w2p + k2 * 32 + 8);
                float4 b3 = *(const float4*)(w2p + k2 * 32 + 12);
                v16bf b;
                b[0]=(__bf16)b0.x; b[1]=(__bf16)b0.y; b[2]=(__bf16)b0.z; b[3]=(__bf16)b0.w;
                b[4]=(__bf16)b1.x; b[5]=(__bf16)b1.y; b[6]=(__bf16)b1.z; b[7]=(__bf16)b1.w;
                b[8]=(__bf16)b2.x; b[9]=(__bf16)b2.y; b[10]=(__bf16)b2.z; b[11]=(__bf16)b2.w;
                b[12]=(__bf16)b3.x; b[13]=(__bf16)b3.y; b[14]=(__bf16)b3.z; b[15]=(__bf16)b3.w;

                acc2[t] = __builtin_amdgcn_wmma_f32_16x16x32_bf16(
                              false, a, false, b, (short)0, acc2[t], false, false);
            }
        }
        __syncthreads();
    }

    // ---------------- epilogue: scale by routing weight, atomic add ----------------
#pragma unroll
    for (int t = 0; t < 8; ++t) {
        int n = (wv * 8 + t) * 16 + nl;
#pragma unroll
        for (int r = 0; r < 8; ++r) {
            int m  = r + half * 8;
            int tk = toks[m];
            if (tk >= 0)
                atomicAdd(&out[(size_t)tk * HH + n], acc2[t][r] * wts[m]);
        }
    }
}

// ---------------------------------------------------------------------------
extern "C" void kernel_launch(void* const* d_in, const int* in_sizes, int n_in,
                              void* d_out, int out_size, void* d_ws, size_t ws_size,
                              hipStream_t stream) {
    const float* x   = (const float*)d_in[0];   // (B,S,H)
    const float* gw  = (const float*)d_in[1];   // (E,H)
    const float* w13 = (const float*)d_in[2];   // (E,2F,H)
    const float* w2  = (const float*)d_in[3];   // (E,H,F)

    float* out    = (float*)d_out;              // T*H floats
    float* logits = out + (size_t)TT * HH;      // then T*E floats

    char* ws = (char*)d_ws;
    int*    counts = (int*)ws;
    int*    etok   = (int*)  (ws + 256);
    float*  ewt    = (float*)(ws + 256 + (size_t)EE * TT * 4);
    __bf16* xbf    = (__bf16*)(ws + 256 + (size_t)EE * TT * 8);

    moe_zero  <<<(TT * HH) / (256 * 4), 256, 0, stream>>>(out, counts);
    moe_router<<<TT / 8, 256, 0, stream>>>(x, gw, logits, xbf, counts, etok, ewt);
    dim3 g(EE, TT / 16);
    moe_ffn   <<<g, 256, 0, stream>>>(xbf, w13, w2, counts, etok, ewt, out);
}
